// SelfAttention_45148696216037
// MI455X (gfx1250) — matrix operand order repaired and checked by reading, not compiled, not run on hardware
//
#include <hip/hip_runtime.h>
#include <hip/hip_bf16.h>

#define HEADS 16
#define DHEAD 64

typedef __attribute__((ext_vector_type(16))) __bf16 v16bf;
typedef __attribute__((ext_vector_type(8)))  float  v8f;

// ---------- CDNA5 async copy helpers (ASYNCcnt path) ----------

// One 16-byte chunk: global -> LDS, bypassing VGPRs. Tracked by ASYNCcnt.
__device__ __forceinline__ void async_g2l_b128(uint32_t lds_addr, uint64_t gaddr) {
    asm volatile("global_load_async_to_lds_b128 %0, %1, off"
                 :: "v"(lds_addr), "v"(gaddr) : "memory");
}

__device__ __forceinline__ void wait_async0() {
#if __has_builtin(__builtin_amdgcn_s_wait_asynccnt)
    __builtin_amdgcn_s_wait_asynccnt(0);
#else
    asm volatile("s_wait_asynccnt 0x0" ::: "memory");
#endif
}

// ---------- WMMA helpers (CDNA5 wave32, 16x16x32 bf16, fp32 accum) ----------

__device__ __forceinline__ v8f wmma_bf16(v16bf a, v16bf b, v8f c) {
    // 8 args: (neg_a, A, neg_b, B, c_mod, C, reuse_a, reuse_b)
    return __builtin_amdgcn_wmma_f32_16x16x32_bf16(false, a, false, b, (short)0, c,
                                                   false, false);
}

// A-matrix 16x32 bf16 fragment layout (contiguous, matches B loaders):
// lanes 0-15: row M=lane, K=0..15 in v[0..15]
// lanes 16-31: row M=lane-16, K=16..31
__device__ __forceinline__ int koffA(int p, int lane) {
    int kbase = (lane >> 4) * 16;
    return kbase + 2 * p;
}

// Load A fragment from row-major bf16 [16 x >=32] tile, leading dim ld.
__device__ __forceinline__ v16bf load_a_frag(const __bf16* __restrict__ src, int ld, int lane) {
    int row = lane & 15;
    v16bf a;
#pragma unroll
    for (int p = 0; p < 8; ++p) {
        int k = koffA(p, lane);
        a[2 * p]     = src[row * ld + k];
        a[2 * p + 1] = src[row * ld + k + 1];
    }
    return a;
}

// B-matrix 32x16 bf16 fragment: lanes 0-15 hold K=0..15 (v[2j]=K=2j), lanes 16-31 K=16..31;
// column n = lane&15.  Here B[k][n] = srcT[n*ld + k] (transposed source, i.e. x @ W^T / Q K^T).
__device__ __forceinline__ v16bf load_bT_frag_bf16(const __bf16* __restrict__ srcT, int ld, int lane) {
    int n  = lane & 15;
    int kb = (lane >> 4) * 16;
    v16bf b;
#pragma unroll
    for (int j = 0; j < 8; ++j) {
        b[2 * j]     = srcT[n * ld + kb + 2 * j];
        b[2 * j + 1] = srcT[n * ld + kb + 2 * j + 1];
    }
    return b;
}

// B fragment from natural-order source: B[k][n] = src[k*ld + n] (for P @ V).
__device__ __forceinline__ v16bf load_b_frag_bf16(const __bf16* __restrict__ src, int ld, int lane) {
    int n  = lane & 15;
    int kb = (lane >> 4) * 16;
    v16bf b;
#pragma unroll
    for (int j = 0; j < 8; ++j) {
        b[2 * j]     = src[(kb + 2 * j) * ld + n];
        b[2 * j + 1] = src[(kb + 2 * j + 1) * ld + n];
    }
    return b;
}

// ---------- Kernel 0: fp32 -> bf16 elementwise (for Wfc; 2 MB result lives in L2) ----------
__global__ __launch_bounds__(256) void cvt_bf16_kernel(
    const float* __restrict__ src, __bf16* __restrict__ dst, int n)
{
    int i = (blockIdx.x * 256 + threadIdx.x) * 4;
    if (i + 3 < n) {
        float4 f = *(const float4*)(src + i);
        dst[i + 0] = (__bf16)f.x;
        dst[i + 1] = (__bf16)f.y;
        dst[i + 2] = (__bf16)f.z;
        dst[i + 3] = (__bf16)f.w;
    }
}

// ---------- Kernel 1: per-head projection  y = x_head @ W^T  (fp32 in, bf16 out) ----------
// out layout: [N, H, L, D] head-major for the attention kernel.
__global__ __launch_bounds__(256) void proj_bf16_kernel(
    const float* __restrict__ x,   // [N, L, E]
    const float* __restrict__ W,   // [D, D]
    __bf16* __restrict__ out,      // [N, H, L, D]
    int L)
{
    const int E = HEADS * DHEAD;
    __shared__ __bf16 Wbf[DHEAD * DHEAD];  // W[n][k]
    __shared__ __bf16 Xbf[32 * DHEAD];     // 32 rows of this head's slice

    int h   = blockIdx.y;
    int nb  = blockIdx.z;
    int r0  = blockIdx.x * 32;
    int tid = threadIdx.x;

    for (int i = tid; i < DHEAD * DHEAD; i += 256) Wbf[i] = (__bf16)W[i];
    const float* xs = x + (size_t)nb * L * E + (size_t)r0 * E + h * DHEAD;
    for (int i = tid; i < 32 * DHEAD; i += 256) {
        int rr = i >> 6, cc = i & 63;
        Xbf[i] = (__bf16)xs[(size_t)rr * E + cc];
    }
    __syncthreads();

    int wave = tid >> 5, lane = tid & 31;
    int rowtile = wave >> 2, coltile = wave & 3;  // 2 row tiles x 4 col tiles = 32x64

    v8f acc = {};
#pragma unroll
    for (int kb = 0; kb < DHEAD; kb += 32) {
        v16bf a = load_a_frag(&Xbf[rowtile * 16 * DHEAD + kb], DHEAD, lane);
        v16bf b = load_bT_frag_bf16(&Wbf[coltile * 16 * DHEAD + kb], DHEAD, lane);
        acc = wmma_bf16(a, b, acc);
    }

    // C layout: lane n = lane&15 (col), rows mbase..mbase+7
    int mbase = (lane >> 4) * 8;
    int col   = coltile * 16 + (lane & 15);
    size_t base = (((size_t)nb * HEADS + h) * L + r0 + rowtile * 16 + mbase) * DHEAD;
#pragma unroll
    for (int r = 0; r < 8; ++r)
        out[base + (size_t)r * DHEAD + col] = (__bf16)acc[r];
}

// ---------- Kernel 2: fused flash attention ----------
// Each wave: 16 query rows, online softmax, O accum 16x64 fp32.
// Block (8 waves) shares one (n,h). K/V 32-key tiles double-buffered in LDS via
// GLOBAL_LOAD_ASYNC_TO_LDS_B128 (ASYNCcnt): next tile streams in while WMMAs run.
__global__ __launch_bounds__(256) void flash_attn_kernel(
    const __bf16* __restrict__ Q,   // [N,H,L,D]
    const __bf16* __restrict__ K,   // [N,H,L,D]
    const __bf16* __restrict__ V,   // [N,H,L,D]
    const int*    __restrict__ mask,// [N, L]
    __bf16* __restrict__ out,       // [N, L, E]
    int L)
{
    const int E = HEADS * DHEAD;
    __shared__ __bf16 KV[2][2][32 * DHEAD];   // [buf][K/V], 16 KB
    __shared__ __bf16 Plds[8][16 * 32];       // 8 KB, wave-private P re-layout scratch

    int h = blockIdx.y, nb = blockIdx.z;
    int tid  = threadIdx.x;
    int wave = tid >> 5, lane = tid & 31;
    int qtile = blockIdx.x * 8 + wave;

    const __bf16* Qh = Q + (((size_t)nb * HEADS + h) * L) * DHEAD;
    const __bf16* Kh = K + (((size_t)nb * HEADS + h) * L) * DHEAD;
    const __bf16* Vh = V + (((size_t)nb * HEADS + h) * L) * DHEAD;
    const int*  mrow = mask + (size_t)nb * L;

    const float scale = 0.03125f;  // 1/sqrt(E) = 1/32

    v8f o0 = {}, o1 = {}, o2 = {}, o3 = {};   // O: 16 rows x 64 cols fp32
    float mstat[8], lstat[8];
#pragma unroll
    for (int r = 0; r < 8; ++r) { mstat[r] = -1e30f; lstat[r] = 0.f; }

    // Q fragments are loop-invariant: preload both K-halves (d 0..31, 32..63)
    v16bf qa0 = load_a_frag(Qh + (size_t)qtile * 16 * DHEAD + 0,  DHEAD, lane);
    v16bf qa1 = load_a_frag(Qh + (size_t)qtile * 16 * DHEAD + 32, DHEAD, lane);

    int mbase = (lane >> 4) * 8;
    int ncol  = lane & 15;

    // per-thread 16B chunk offset inside a 4 KB tile (32 keys x 64 bf16)
    const int chunk = tid * 16;
    uint32_t ldsK[2], ldsV[2];
    ldsK[0] = (uint32_t)(uintptr_t)&KV[0][0][0] + chunk;
    ldsV[0] = (uint32_t)(uintptr_t)&KV[0][1][0] + chunk;
    ldsK[1] = (uint32_t)(uintptr_t)&KV[1][0][0] + chunk;
    ldsV[1] = (uint32_t)(uintptr_t)&KV[1][1][0] + chunk;

    // prime buffer 0 with keys/values [0, 32)
    async_g2l_b128(ldsK[0], (uint64_t)(uintptr_t)Kh + chunk);
    async_g2l_b128(ldsV[0], (uint64_t)(uintptr_t)Vh + chunk);
    wait_async0();
    __syncthreads();

    for (int kb = 0; kb < L; kb += 32) {
        int cur = (kb >> 5) & 1;
        if (kb + 32 < L) {  // prefetch next 32-key tile into the other buffer
            size_t nb16 = (size_t)(kb + 32) * DHEAD * sizeof(__bf16);
            async_g2l_b128(ldsK[cur ^ 1], (uint64_t)(uintptr_t)Kh + nb16 + chunk);
            async_g2l_b128(ldsV[cur ^ 1], (uint64_t)(uintptr_t)Vh + nb16 + chunk);
        }
        const __bf16* Kt = &KV[cur][0][0];
        const __bf16* Vt = &KV[cur][1][0];

        // S = Q * K^T for 16 queries x 32 keys (two 16x16 accumulators)
        v8f s0 = {}, s1 = {};
        s0 = wmma_bf16(qa0, load_bT_frag_bf16(&Kt[0 * DHEAD + 0],  DHEAD, lane), s0);
        s0 = wmma_bf16(qa1, load_bT_frag_bf16(&Kt[0 * DHEAD + 32], DHEAD, lane), s0);
        s1 = wmma_bf16(qa0, load_bT_frag_bf16(&Kt[16 * DHEAD + 0],  DHEAD, lane), s1);
        s1 = wmma_bf16(qa1, load_bT_frag_bf16(&Kt[16 * DHEAD + 32], DHEAD, lane), s1);

        // scale + mask (mask broadcast over heads/queries; key index = column)
        bool mk0 = mrow[kb + ncol] != 0;
        bool mk1 = mrow[kb + 16 + ncol] != 0;
        float p0[8], p1[8];
#pragma unroll
        for (int r = 0; r < 8; ++r) {
            p0[r] = mk0 ? s0[r] * scale : -1e30f;
            p1[r] = mk1 ? s1[r] * scale : -1e30f;
        }

        // online softmax: row reductions across the 16 lanes of each half-wave
#pragma unroll
        for (int r = 0; r < 8; ++r) {
            float t = fmaxf(p0[r], p1[r]);
#pragma unroll
            for (int off = 1; off < 16; off <<= 1) t = fmaxf(t, __shfl_xor(t, off, 32));
            float newm = fmaxf(mstat[r], t);
            float corr = __expf(mstat[r] - newm);
            float e0 = __expf(p0[r] - newm);
            float e1 = __expf(p1[r] - newm);
            float rs = e0 + e1;
#pragma unroll
            for (int off = 1; off < 16; off <<= 1) rs += __shfl_xor(rs, off, 32);
            lstat[r] = lstat[r] * corr + rs;
            mstat[r] = newm;
            o0[r] *= corr; o1[r] *= corr; o2[r] *= corr; o3[r] *= corr;
            p0[r] = e0; p1[r] = e1;
        }

        // re-fragment P: C layout -> A layout via wave-private LDS (in-order per wave)
        __bf16* Pw = &Plds[wave][0];
#pragma unroll
        for (int r = 0; r < 8; ++r) {
            Pw[(mbase + r) * 32 + ncol]      = (__bf16)p0[r];
            Pw[(mbase + r) * 32 + ncol + 16] = (__bf16)p1[r];
        }
        v16bf pa = load_a_frag(Pw, 32, lane);

        // O += P (16x32) * V (32x64), 4 column tiles
        o0 = wmma_bf16(pa, load_b_frag_bf16(&Vt[0],  DHEAD, lane), o0);
        o1 = wmma_bf16(pa, load_b_frag_bf16(&Vt[16], DHEAD, lane), o1);
        o2 = wmma_bf16(pa, load_b_frag_bf16(&Vt[32], DHEAD, lane), o2);
        o3 = wmma_bf16(pa, load_b_frag_bf16(&Vt[48], DHEAD, lane), o3);

        wait_async0();     // prefetched tile landed in LDS
        __syncthreads();   // all waves done reading cur, writes visible
    }

    // normalize and write to [N, L, E] (interleaved heads) as bf16 for the FC GEMM
#pragma unroll
    for (int r = 0; r < 8; ++r) {
        float inv = 1.0f / lstat[r];
        size_t row  = (size_t)qtile * 16 + mbase + r;
        size_t base = ((size_t)nb * L + row) * E + h * DHEAD;
        out[base + 0  + ncol] = (__bf16)(o0[r] * inv);
        out[base + 16 + ncol] = (__bf16)(o1[r] * inv);
        out[base + 32 + ncol] = (__bf16)(o2[r] * inv);
        out[base + 48 + ncol] = (__bf16)(o3[r] * inv);
    }
}

// ---------- Kernel 3: final FC  out = attn @ Wfc^T + b  (all-bf16 operands, fp32 out) ----------
__global__ __launch_bounds__(256) void fc_kernel(
    const __bf16* __restrict__ A,    // [M, E] attention output
    const __bf16* __restrict__ W,    // [E, E] pre-converted Wfc
    const float*  __restrict__ bias, // [E]
    float* __restrict__ out)         // [M, E]
{
    const int E = HEADS * DHEAD;
    __shared__ __bf16 Abf[16 * HEADS * DHEAD];  // 16 x 1024 bf16 = 32 KB

    int mtile = blockIdx.x;
    int ngrp  = blockIdx.y;
    int tid   = threadIdx.x;

    // stage 16-row A strip with async copies (32 KB = 2048 x 16B; 8 chunks/thread)
    {
        uint64_t ga = (uint64_t)(uintptr_t)(A + (size_t)mtile * 16 * E);
        uint32_t la = (uint32_t)(uintptr_t)&Abf[0];
#pragma unroll
        for (int c = 0; c < 8; ++c) {
            int off = (tid + c * 256) * 16;
            async_g2l_b128(la + off, ga + off);
        }
        wait_async0();
        __syncthreads();
    }

    int wave = tid >> 5, lane = tid & 31;
    int ntile = ngrp * 8 + wave;
    const __bf16* Wrow = W + (size_t)ntile * 16 * E;   // 16 rows of Wfc (bf16, L2-resident)

    v8f acc = {};
    for (int kb = 0; kb < E; kb += 32) {
        v16bf a = load_a_frag(&Abf[kb], E, lane);
        v16bf b = load_bT_frag_bf16(Wrow + kb, E, lane);
        acc = wmma_bf16(a, b, acc);
    }

    int mbase = (lane >> 4) * 8, ncol = lane & 15;
    int colg  = ntile * 16 + ncol;
    float bv  = bias[colg];
#pragma unroll
    for (int r = 0; r < 8; ++r)
        out[((size_t)mtile * 16 + mbase + r) * E + colg] = acc[r] + bv;
}

// ---------- host launch ----------
extern "C" void kernel_launch(void* const* d_in, const int* in_sizes, int n_in,
                              void* d_out, int out_size, void* d_ws, size_t ws_size,
                              hipStream_t stream) {
    const float* query  = (const float*)d_in[0];
    const float* keys   = (const float*)d_in[1];
    const float* values = (const float*)d_in[2];
    const int*   mask   = (const int*)d_in[3];
    const float* Wq     = (const float*)d_in[4];
    const float* Wk     = (const float*)d_in[5];
    const float* Wv     = (const float*)d_in[6];
    const float* Wfc    = (const float*)d_in[7];
    const float* bfc    = (const float*)d_in[8];
    float* out = (float*)d_out;

    const int E = HEADS * DHEAD;   // 1024
    const int L = 1024;
    const int N = in_sizes[0] / (L * E);   // 8

    size_t nhld = (size_t)N * HEADS * L * DHEAD;   // == N*L*E
    __bf16* qb  = (__bf16*)d_ws;
    __bf16* kb  = qb + nhld;
    __bf16* vb  = kb + nhld;
    __bf16* ab  = vb + nhld;        // attention output, [N, L, E]
    __bf16* wfb = ab + nhld;        // Wfc in bf16, [E, E]

    dim3 blk(256);

    cvt_bf16_kernel<<<dim3((E * E) / 1024), blk, 0, stream>>>(Wfc, wfb, E * E);

    dim3 gProj(L / 32, HEADS, N);
    proj_bf16_kernel<<<gProj, blk, 0, stream>>>(query,  Wq, qb, L);
    proj_bf16_kernel<<<gProj, blk, 0, stream>>>(keys,   Wk, kb, L);
    proj_bf16_kernel<<<gProj, blk, 0, stream>>>(values, Wv, vb, L);

    dim3 gAttn(L / 16 / 8, HEADS, N);  // 8 waves per block, 16 query rows per wave
    flash_attn_kernel<<<gAttn, blk, 0, stream>>>(qb, kb, vb, mask, ab, L);

    int M = N * L;
    dim3 gFc(M / 16, E / 16 / 8);
    fc_kernel<<<gFc, blk, 0, stream>>>(ab, wfb, bfc, out);
}
